// PolicyHead_70308614635654
// MI455X (gfx1250) — compile-verified, matrix operand order checked by reading, not run on hardware
//
#include <hip/hip_runtime.h>
#include <hip/hip_bf16.h>
#include <math.h>

// ---------------------------------------------------------------------------
// Fully fused policy head for MI455X (gfx1250, wave32, WMMA).
// One workgroup (256 threads = 8 waves) per batch sample.
//   Phase 0: stage X[361,192] f32->bf16 into LDS (coalesced, unroll-8, NT
//            loads), stage W[64,192].
//   Phase 1: GEMM Y[368,64] = X * W^T via v_wmma_f32_16x16x32_bf16.
//            nt = wave&3 (wave-uniform): waves 0,1,4,5 -> "P" channels (kept),
//            waves 2,3,6,7 -> "G" channels (masked gelu + pooling).
//            12 explicit ds_load_b128 (volatile asm, pinned up front) with
//            per-pair non-volatile tie-asms -> waitcnt pass emits staggered
//            partial s_wait_dscnt per WMMA, overlapping LDS latency with the
//            matrix pipe.
//   Phase 2: pool vector [96] -> linear -> g[32]
//   Phase 3: per position: gelu((Yp + g + beta2)*mask) . w2p  -> z[361]
//   Phase 4: log_softmax over 361, write out (NT stores).
// ---------------------------------------------------------------------------

#define CIN   192
#define HW    361
#define MP    368          // 23 tiles of 16 rows
#define MT    23
#define XS_ST 200          // bf16 stride: 400B rows, 16B aligned, conflict-friendly
#define YP_ST 36           // f32 stride: 144B rows, 16B aligned
#define P1    32
#define G1    32

typedef __attribute__((ext_vector_type(16))) __bf16 v16bf;
typedef __attribute__((ext_vector_type(8)))  __bf16 v8bf;
typedef __attribute__((ext_vector_type(8)))  float  v8f;
typedef __attribute__((ext_vector_type(4)))  float  v4f;

__device__ __forceinline__ float gelu_exact(float x) {
    return 0.5f * x * (1.0f + erff(x * 0.70710678118654752440f));
}
__device__ __forceinline__ unsigned fkey(float f) {        // monotone float->uint
    unsigned b = __float_as_uint(f);
    return (b & 0x80000000u) ? ~b : (b | 0x80000000u);
}
__device__ __forceinline__ float funkey(unsigned k) {
    return (k & 0x80000000u) ? __uint_as_float(k ^ 0x80000000u)
                             : __uint_as_float(~k);
}

// LDS b128 load via explicit ds instruction (addr = LDS byte offset).
// Volatile: mutual order pins all 12 issues before the WMMA chain.
#define LDS_LD_B128(dst, addr, off) \
    asm volatile("ds_load_b128 %0, %1 offset:" #off : "=v"(dst) : "v"(addr))
// Non-volatile zero-instruction tie: floats to just before its consumer WMMA
// (data deps only), so the waitcnt pass emits a partial wait for this pair.
#define PAIR_TIE(x0, x1) asm("" : "+v"(x0), "+v"(x1))

__global__ __launch_bounds__(256)
void policy_head_fused(const float* __restrict__ x,        // [B,192,19,19]
                       const float* __restrict__ mask,     // [B,1,19,19]
                       const float* __restrict__ mask_sum, // [B]
                       const float* __restrict__ w1p,      // [32,192]
                       const float* __restrict__ b1p,      // [32]
                       const float* __restrict__ w1g,      // [32,192]
                       const float* __restrict__ b1g,      // [32]
                       const float* __restrict__ beta_g,   // [32]
                       const float* __restrict__ w_lin,    // [32,96]
                       const float* __restrict__ b_lin,    // [32]
                       const float* __restrict__ beta2,    // [32]
                       const float* __restrict__ w2p,      // [32]
                       const float* __restrict__ b2p,      // [1]
                       float* __restrict__ out)            // [B,361]
{
    __shared__ __align__(16) __bf16 Xs[MP * XS_ST];   // 147200 B
    __shared__ __align__(16) __bf16 Wsm[64 * XS_ST];  //  25600 B
    __shared__ __align__(16) float  Yp[MP * YP_ST];   //  52992 B
    __shared__ __align__(16) float  msk[MP];
    __shared__ float    gsum[G1];
    __shared__ unsigned gmaxb[G1];
    __shared__ float    pool[3 * G1];
    __shared__ float    gvec[P1];
    __shared__ float    sb[5 * 32];   // b1p | b1g | beta_g | beta2 | w2p
    __shared__ float    zbuf[MP];
    __shared__ float    red[256];

    const int b   = blockIdx.x;
    const int tid = threadIdx.x;

    // ---- Phase 0: staging -------------------------------------------------
    if (tid < 32) {
        sb[tid]       = b1p[tid];
        sb[32 + tid]  = b1g[tid];
        sb[64 + tid]  = beta_g[tid];
        sb[96 + tid]  = beta2[tid];
        sb[128 + tid] = w2p[tid];
        gsum[tid]  = 0.0f;
        gmaxb[tid] = 0u;                       // smallest key
    }
    for (int p = tid; p < MP; p += 256)
        msk[p] = (p < HW) ? mask[b * HW + p] : 0.0f;

    for (int i = tid; i < 64 * CIN; i += 256) {
        int o = i / CIN, c = i % CIN;
        float wv = (o < 32) ? w1p[o * CIN + c] : w1g[(o - 32) * CIN + c];
        Wsm[o * XS_ST + c] = (__bf16)wv;
    }
    // Stream X: coalesced over contiguous position dim; unroll-8 batches the
    // global loads (LDS stores don't alias them); NT hint = zero-reuse data.
    const float* xb = x + (size_t)b * (CIN * HW);
#pragma unroll 8
    for (int i = tid; i < CIN * HW; i += 256) {
        int c = i / HW, p = i % HW;
        Xs[p * XS_ST + c] = (__bf16)__builtin_nontemporal_load(&xb[i]);
    }
    for (int i = tid; i < (MP - HW) * XS_ST; i += 256)
        Xs[HW * XS_ST + i] = (__bf16)0.0f;     // zero pad rows 361..367
    __syncthreads();

    // ---- Phase 1: WMMA GEMM + fused epilogue ------------------------------
    const int wave = tid >> 5;
    const int lane = tid & 31;
    const int lrow = lane & 15;
    const int half = lane >> 4;                // 0: K chunk offset 0, 1: +8

    // nt is wave-uniform: wave&3. mt = (wave>>2) + 2k.
    const int nt = wave & 3;
    const int n0 = nt * 16;
    const bool is_p = (nt < 2);                // uniform per wave

    // Hoist the 6 B fragments (weights) into registers for this wave's nt.
    const __bf16* brow = &Wsm[(n0 + lrow) * XS_ST + 8 * half];
    v16bf bfr[6];
#pragma unroll
    for (int kb = 0; kb < 6; ++kb) {
        v8bf blo = *(const v8bf*)(brow + kb * 32);
        v8bf bhi = *(const v8bf*)(brow + kb * 32 + 16);
        bfr[kb] = __builtin_shufflevector(blo, bhi,
            0,1,2,3,4,5,6,7,8,9,10,11,12,13,14,15);
    }

    // Loop-invariant epilogue constants (hoisted: sb[] LDS loads once).
    const int   nch    = n0 + lrow;            // D layout: N = lane%16
    const int   cg     = nch & 31;
    const float bias_p = sb[nch];              // P path: b1p[nch]
    const float bs_g   = sb[32 + cg] + sb[64 + cg];  // G path: b1g + beta_g

    for (int mt = (wave >> 2); mt < MT; mt += 2) {
        const int m0 = mt * 16;
        // LDS byte address of this lane's A-row chunk base.
        const unsigned abase =
            (unsigned)(uintptr_t)(&Xs[(m0 + lrow) * XS_ST + 8 * half]);

        // Issue ALL 12 ds_load_b128 up front; chunk kb at bytes kb*64, kb*64+32.
        v8bf a0,a1,a2,a3,a4,a5,a6,a7,a8,a9,a10,a11;
        LDS_LD_B128(a0,  abase, 0);
        LDS_LD_B128(a1,  abase, 32);
        LDS_LD_B128(a2,  abase, 64);
        LDS_LD_B128(a3,  abase, 96);
        LDS_LD_B128(a4,  abase, 128);
        LDS_LD_B128(a5,  abase, 160);
        LDS_LD_B128(a6,  abase, 192);
        LDS_LD_B128(a7,  abase, 224);
        LDS_LD_B128(a8,  abase, 256);
        LDS_LD_B128(a9,  abase, 288);
        LDS_LD_B128(a10, abase, 320);
        LDS_LD_B128(a11, abase, 352);

        v8f acc = {};
        PAIR_TIE(a0, a1);
        acc = __builtin_amdgcn_wmma_f32_16x16x32_bf16(false,
            __builtin_shufflevector(a0, a1, 0,1,2,3,4,5,6,7,8,9,10,11,12,13,14,15),
            false, bfr[0], (short)0, acc, false, false);
        PAIR_TIE(a2, a3);
        acc = __builtin_amdgcn_wmma_f32_16x16x32_bf16(false,
            __builtin_shufflevector(a2, a3, 0,1,2,3,4,5,6,7,8,9,10,11,12,13,14,15),
            false, bfr[1], (short)0, acc, false, false);
        PAIR_TIE(a4, a5);
        acc = __builtin_amdgcn_wmma_f32_16x16x32_bf16(false,
            __builtin_shufflevector(a4, a5, 0,1,2,3,4,5,6,7,8,9,10,11,12,13,14,15),
            false, bfr[2], (short)0, acc, false, false);
        PAIR_TIE(a6, a7);
        acc = __builtin_amdgcn_wmma_f32_16x16x32_bf16(false,
            __builtin_shufflevector(a6, a7, 0,1,2,3,4,5,6,7,8,9,10,11,12,13,14,15),
            false, bfr[3], (short)0, acc, false, false);
        PAIR_TIE(a8, a9);
        acc = __builtin_amdgcn_wmma_f32_16x16x32_bf16(false,
            __builtin_shufflevector(a8, a9, 0,1,2,3,4,5,6,7,8,9,10,11,12,13,14,15),
            false, bfr[4], (short)0, acc, false, false);
        PAIR_TIE(a10, a11);
        acc = __builtin_amdgcn_wmma_f32_16x16x32_bf16(false,
            __builtin_shufflevector(a10, a11, 0,1,2,3,4,5,6,7,8,9,10,11,12,13,14,15),
            false, bfr[5], (short)0, acc, false, false);

        if (is_p) {                            // "P" channels -> keep for stage 2
#pragma unroll
            for (int r = 0; r < 8; ++r) {
                const int M = m0 + r + 8 * half;
                Yp[M * YP_ST + nch] = acc[r] + bias_p;
            }
        } else {                               // "G" channels -> gelu + pool
            // This lane's 8 rows are consecutive: msk[m0+8*half .. +7].
            const v4f* mp = (const v4f*)&msk[m0 + 8 * half];
            const v4f mk0 = mp[0], mk1 = mp[1];
            float lsum = 0.0f, lmax = -3.0e38f;
#pragma unroll
            for (int r = 0; r < 8; ++r) {
                const float mm = (r < 4) ? mk0[r & 3] : mk1[r & 3];
                const float ge = gelu_exact((acc[r] + bs_g) * mm);
                lsum += ge;                     // pad rows: mm=0 -> ge=0
                lmax  = fmaxf(lmax, ge + mm - 1.0f);  // pad candidate -1 is safe
            }
            // lanes l and l+16 hold the same channel, different rows
            lsum += __shfl_xor(lsum, 16, 32);
            lmax  = fmaxf(lmax, __shfl_xor(lmax, 16, 32));
            if (half == 0) {
                atomicAdd(&gsum[cg], lsum);
                atomicMax(&gmaxb[cg], fkey(lmax));
            }
        }
    }
    __syncthreads();

    // ---- Phase 2: pool vector + 96->32 linear -----------------------------
    if (tid < G1) {
        const float ms   = mask_sum[b];
        const float mean = gsum[tid] / ms;
        const float off  = (sqrtf(ms) - 28.0f) * 0.1f;
        pool[tid]          = mean;
        pool[G1 + tid]     = mean * off;
        pool[2 * G1 + tid] = funkey(gmaxb[tid]);
    }
    __syncthreads();
    if (tid < P1) {
        float a = b_lin[tid];
        const float* wl = w_lin + tid * (3 * G1);
        for (int k = 0; k < 3 * G1; ++k) a += pool[k] * wl[k];
        gvec[tid] = a;
    }
    __syncthreads();

    // ---- Phase 3: second gelu + 32->1 conv --------------------------------
    const float b2 = b2p[0];
    for (int p = tid; p < HW; p += 256) {
        const float mm = msk[p];
        const v4f* yr4 = (const v4f*)&Yp[p * YP_ST];   // 16B-aligned rows
        v4f yv[8];
#pragma unroll
        for (int q = 0; q < 8; ++q) yv[q] = yr4[q];
        float a = 0.0f;
#pragma unroll
        for (int c = 0; c < P1; ++c) {
            const float v = (yv[c >> 2][c & 3] + gvec[c] + sb[96 + c]) * mm;
            a += sb[128 + c] * gelu_exact(v);
        }
        zbuf[p] = a + b2 - (1.0f - mm) * 5000.0f;
    }
    __syncthreads();

    // ---- Phase 4: log_softmax over 361 ------------------------------------
    float lm = -3.0e38f;
    for (int p = tid; p < HW; p += 256) lm = fmaxf(lm, zbuf[p]);
    red[tid] = lm;
    __syncthreads();
    for (int s = 128; s > 0; s >>= 1) {
        if (tid < s) red[tid] = fmaxf(red[tid], red[tid + s]);
        __syncthreads();
    }
    const float zmax = red[0];
    __syncthreads();
    float ls = 0.0f;
    for (int p = tid; p < HW; p += 256) ls += expf(zbuf[p] - zmax);
    red[tid] = ls;
    __syncthreads();
    for (int s = 128; s > 0; s >>= 1) {
        if (tid < s) red[tid] += red[tid + s];
        __syncthreads();
    }
    const float lse = logf(red[0]) + zmax;
    for (int p = tid; p < HW; p += 256)
        __builtin_nontemporal_store(zbuf[p] - lse, &out[(size_t)b * HW + p]);
}

extern "C" void kernel_launch(void* const* d_in, const int* in_sizes, int n_in,
                              void* d_out, int out_size, void* d_ws, size_t ws_size,
                              hipStream_t stream) {
    const float* x        = (const float*)d_in[0];
    const float* mask     = (const float*)d_in[1];
    const float* mask_sum = (const float*)d_in[2];
    const float* w1p      = (const float*)d_in[3];
    const float* b1p      = (const float*)d_in[4];
    const float* w1g      = (const float*)d_in[5];
    const float* b1g      = (const float*)d_in[6];
    const float* beta_g   = (const float*)d_in[7];
    const float* w_lin    = (const float*)d_in[8];
    const float* b_lin    = (const float*)d_in[9];
    const float* beta2    = (const float*)d_in[10];
    const float* w2p      = (const float*)d_in[11];
    const float* b2p      = (const float*)d_in[12];
    float*       out      = (float*)d_out;

    const int B = in_sizes[0] / (CIN * HW);    // 2048
    policy_head_fused<<<dim3(B), dim3(256), 0, stream>>>(
        x, mask, mask_sum, w1p, b1p, w1g, b1g, beta_g,
        w_lin, b_lin, beta2, w2p, b2p, out);
}
